// Extractor_77464030150790
// MI455X (gfx1250) — matrix operand order repaired and checked by reading, not compiled, not run on hardware
//
#include <hip/hip_runtime.h>
#include <hip/hip_bf16.h>
#include <math.h>

typedef __attribute__((ext_vector_type(16))) __bf16 v16bf;
typedef __attribute__((ext_vector_type(8)))  float  v8f;
typedef __attribute__((ext_vector_type(4)))  unsigned int tdm_u32x4;
typedef __attribute__((ext_vector_type(8)))  int          tdm_i32x8;
typedef __attribute__((ext_vector_type(4)))  int          tdm_i32x4;

#define LQ    4096
#define BATCH 4
#define CDIM  768
#define NHEAD 12
#define HD    64

#if defined(__has_builtin)
#if __has_builtin(__builtin_amdgcn_tensor_load_to_lds) && __has_builtin(__builtin_amdgcn_s_wait_tensorcnt)
#define USE_TDM 1
#endif
#endif
#ifndef USE_TDM
#define USE_TDM 0
#endif

__device__ __forceinline__ unsigned short f2bf(float f) {
    unsigned int u = __float_as_uint(f);
    u += 0x7FFFu + ((u >> 16) & 1u);           // round-to-nearest-even
    return (unsigned short)(u >> 16);
}
__device__ __forceinline__ float bf2f(unsigned short h) {
    return __uint_as_float(((unsigned int)h) << 16);
}

// ---------------------------------------------------------------- TDM helper
// D# (ISA ch.8): 2-D tile [tile_dim1=128 rows x tile_dim0=32 elems], bf16,
// global_addr = tile start (OOB checks tile-relative via remaining extents),
// LDS pad 8 DWORDs every 16 DWORDs (matches 32+16 ushort row stride).
__device__ __forceinline__ void tdm_load_tile(const unsigned short* gptr,
                                              unsigned lds_off,
                                              int dim0_rem, int dim1_rem,
                                              int stride) {
#if USE_TDM
    unsigned long long ga = (unsigned long long)gptr;
    tdm_u32x4 g0;
    g0[0] = 1u;                                        // count=1, user desc
    g0[1] = lds_off;                                   // lds_addr
    g0[2] = (unsigned)(ga & 0xFFFFFFFFu);              // global_addr lo
    g0[3] = (unsigned)((ga >> 32) & 0x01FFFFFFu) | 0x80000000u;  // ga hi | type=2
    tdm_i32x8 g1;
    g1[0] = (int)((1u << 16)      // data_size = 2 bytes
                | (1u << 20)      // pad_enable
                | (3u << 22)      // pad_interval = 16 DWORDs
                | (7u << 25));    // pad_amount   = 8 DWORDs
    g1[1] = (int)(((unsigned)dim0_rem & 0xFFFFu) << 16);             // tensor_dim0 lo16
    g1[2] = (int)((((unsigned)dim0_rem >> 16) & 0xFFFFu)
                | (((unsigned)dim1_rem & 0xFFFFu) << 16));           // dim0 hi | dim1 lo
    g1[3] = (int)((((unsigned)dim1_rem >> 16) & 0xFFFFu) | (32u << 16)); // dim1 hi | tile_dim0=32
    g1[4] = 128;                                                     // tile_dim1=128
    g1[5] = stride;                                                  // tensor_dim0_stride lo32
    g1[6] = 0;
    g1[7] = 0;
    tdm_i32x4 g2, g3;
    for (int i = 0; i < 4; ++i) { g2[i] = 0; g3[i] = 0; }
#if __clang_major__ >= 23
    tdm_i32x8 gz;
    for (int i = 0; i < 8; ++i) gz[i] = 0;
    __builtin_amdgcn_tensor_load_to_lds(g0, g1, g2, g3, gz, 0);
#else
    __builtin_amdgcn_tensor_load_to_lds(g0, g1, g2, g3, 0);
#endif
#else
    (void)gptr; (void)lds_off; (void)dim0_rem; (void)dim1_rem; (void)stride;
#endif
}

// ---------------------------------------------------------------- weight prep
// Transpose-convert: src [K,N] f32 -> dst [N,K] bf16
__global__ void k_cvt_transpose(const float* __restrict__ src,
                                unsigned short* __restrict__ dst, int K, int N) {
    int i = blockIdx.x * blockDim.x + threadIdx.x;
    if (i >= K * N) return;
    int k = i / N, n = i % N;
    dst[(size_t)n * K + k] = f2bf(src[i]);
}

// Combined [Woff|Watt] transposed: dst [144, 768] bf16
__global__ void k_pack_offatt_w(const float* __restrict__ Woff,   // [768,96]
                                const float* __restrict__ Watt,   // [768,48]
                                unsigned short* __restrict__ dst) // [144,768]
{
    int i = blockIdx.x * blockDim.x + threadIdx.x;
    if (i >= 768 * 144) return;
    int k = i / 144, c = i % 144;
    float v = (c < 96) ? Woff[k * 96 + c] : Watt[k * 48 + (c - 96)];
    dst[(size_t)c * 768 + k] = f2bf(v);
}

__global__ void k_pack_offatt_b(const float* __restrict__ boff,
                                const float* __restrict__ batt,
                                float* __restrict__ dst) {
    int i = threadIdx.x;
    if (i < 96)       dst[i] = boff[i];
    else if (i < 144) dst[i] = batt[i - 96];
}

// ---------------------------------------------------------------- layernorm
__global__ __launch_bounds__(256)
void k_layernorm_bf16(const float* __restrict__ x,
                      const float* __restrict__ g,
                      const float* __restrict__ b,
                      unsigned short* __restrict__ y) {
    int row = blockIdx.x;                       // one row (768 cols) per block
    const float* xr = x + (size_t)row * CDIM;
    float lx[3];
    float s = 0.f;
    for (int i = 0; i < 3; ++i) { lx[i] = xr[threadIdx.x + 256 * i]; s += lx[i]; }
    __shared__ float red[256];
    red[threadIdx.x] = s; __syncthreads();
    for (int off = 128; off > 0; off >>= 1) {
        if (threadIdx.x < off) red[threadIdx.x] += red[threadIdx.x + off];
        __syncthreads();
    }
    float mean = red[0] * (1.0f / CDIM);
    __syncthreads();
    float vs = 0.f;
    for (int i = 0; i < 3; ++i) { float d = lx[i] - mean; vs += d * d; }
    red[threadIdx.x] = vs; __syncthreads();
    for (int off = 128; off > 0; off >>= 1) {
        if (threadIdx.x < off) red[threadIdx.x] += red[threadIdx.x + off];
        __syncthreads();
    }
    float rstd = rsqrtf(red[0] * (1.0f / CDIM) + 1e-5f);
    unsigned short* yr = y + (size_t)row * CDIM;
    for (int i = 0; i < 3; ++i) {
        int c = threadIdx.x + 256 * i;
        yr[c] = f2bf((lx[i] - mean) * rstd * g[c] + b[c]);
    }
}

// ---------------------------------------------------------------- WMMA GEMM
// C[M,N] = epi(A[M,K](bf16) @ Bt[N,K](bf16, pre-transposed) + bias [+res])
// BM=128 BN=128 BK=32; 8 waves in 4(M)x2(N); wave tile 32x64 = 8 WMMA/k-step.
// TDM path: double-buffered LDS tiles, TDM for tile k+1 overlaps WMMA on
// tile k; s_wait_tensorcnt(2) waits only for the older tile pair.
template <bool OUT_BF16, bool GELU_ACT, bool HAS_RES, bool FULL_N>
__global__ __launch_bounds__(256)
void k_gemm_bf16(const unsigned short* __restrict__ A,
                 const unsigned short* __restrict__ Bt,
                 const float* __restrict__ bias,
                 const float* __restrict__ res,
                 void* __restrict__ out,
                 int M, int N, int K) {
    constexpr int BM = 128, BK = 32, PAD = 16, LDW = BK + PAD;
    constexpr unsigned TILE_BYTES = BM * LDW * 2;   // 12288
    __shared__ unsigned short lds_all[2][2][BM][LDW];  // [buf][A/B][row][k]

    const int t = threadIdx.x;
    const int wave = t >> 5, lane = t & 31;
    const int lane15 = lane & 15, half = lane >> 4;
    const int wm = wave >> 1, wn = wave & 1;
    const int m0 = blockIdx.y * BM;
    const int n0 = blockIdx.x * BM;

    v8f acc[2][4];
    for (int i = 0; i < 2; ++i)
        for (int j = 0; j < 4; ++j)
            for (int e = 0; e < 8; ++e) acc[i][j][e] = 0.f;

    // ---- per-tile compute: fragments (ISA 7.12.2 layouts) + 8 WMMA
    auto compute_tile = [&](const unsigned short (*As)[LDW],
                            const unsigned short (*Bs)[LDW]) {
        union U { uint4 u[2]; v16bf v; };
        U af[2];
        for (int i = 0; i < 2; ++i) {
            int r = wm * 32 + i * 16 + lane15;
            af[i].u[0] = *(const uint4*)&As[r][half * 8];
            af[i].u[1] = *(const uint4*)&As[r][16 + half * 8];
        }
        for (int j = 0; j < 4; ++j) {
            U bfr;
            int c = wn * 64 + j * 16 + lane15;
            bfr.u[0] = *(const uint4*)&Bs[c][half * 8];
            bfr.u[1] = *(const uint4*)&Bs[c][16 + half * 8];
            for (int i = 0; i < 2; ++i)
                acc[i][j] = __builtin_amdgcn_wmma_f32_16x16x32_bf16(
                    false, af[i].v, false, bfr.v, (short)0, acc[i][j], false, false);
        }
    };

#if USE_TDM
    // prologue: stage first tile pair into buffer 0
    if (wave == 0) {
        tdm_load_tile(A  + (size_t)m0 * K, 0u,         K, M - m0, K);
        tdm_load_tile(Bt + (size_t)n0 * K, TILE_BYTES, K, N - n0, K);
    }
    int buf = 0;
    for (int k0 = 0; k0 < K; k0 += BK) {
        if (wave == 0) {
            int kn = k0 + BK;
            if (kn < K) {
                unsigned boff = (unsigned)(buf ^ 1) * 2u * TILE_BYTES;
                tdm_load_tile(A  + (size_t)m0 * K + kn, boff,              K - kn, M - m0, K);
                tdm_load_tile(Bt + (size_t)n0 * K + kn, boff + TILE_BYTES, K - kn, N - n0, K);
                __builtin_amdgcn_s_wait_tensorcnt(2);  // older pair landed
            } else {
                __builtin_amdgcn_s_wait_tensorcnt(0);  // last pair landed
            }
        }
        __syncthreads();
        compute_tile(lds_all[buf][0], lds_all[buf][1]);
        __syncthreads();   // all waves done reading before buf is overwritten
        buf ^= 1;
    }
#else
    for (int k0 = 0; k0 < K; k0 += BK) {
        {   // A tile [128 rows x 32 k]
            int r = t >> 1, cc = (t & 1) * 16;
            *(uint4*)&lds_all[0][0][r][cc]     = *(const uint4*)(A + (size_t)(m0 + r) * K + k0 + cc);
            *(uint4*)&lds_all[0][0][r][cc + 8] = *(const uint4*)(A + (size_t)(m0 + r) * K + k0 + cc + 8);
        }
        {   // B tile [128 n-rows x 32 k] from pre-transposed Bt
            int r = t >> 1, cc = (t & 1) * 16;
            if (FULL_N || n0 + r < N) {
                *(uint4*)&lds_all[0][1][r][cc]     = *(const uint4*)(Bt + (size_t)(n0 + r) * K + k0 + cc);
                *(uint4*)&lds_all[0][1][r][cc + 8] = *(const uint4*)(Bt + (size_t)(n0 + r) * K + k0 + cc + 8);
            } else {
                uint4 z; z.x = z.y = z.z = z.w = 0u;
                *(uint4*)&lds_all[0][1][r][cc]     = z;
                *(uint4*)&lds_all[0][1][r][cc + 8] = z;
            }
        }
        __syncthreads();
        compute_tile(lds_all[0][0], lds_all[0][1]);
        __syncthreads();
    }
#endif

    // ---- epilogue: lane -> N, vgpr e -> M = half*8 + e
    for (int j = 0; j < 4; ++j) {
        int col = n0 + wn * 64 + j * 16 + lane15;
        if (!FULL_N && col >= N) continue;
        float bb = bias[col];
        for (int i = 0; i < 2; ++i)
            for (int e = 0; e < 8; ++e) {
                int row = m0 + wm * 32 + i * 16 + half * 8 + e;
                float v = acc[i][j][e] + bb;
                if (HAS_RES)  v += res[(size_t)row * N + col];
                if (GELU_ACT) v = 0.5f * v * (1.0f + erff(v * 0.70710678118654752f));
                if (OUT_BF16) ((unsigned short*)out)[(size_t)row * N + col] = f2bf(v);
                else          ((float*)out)[(size_t)row * N + col] = v;
            }
    }
}

// ---------------------------------------------------------------- deform sampling
// grid (Lq, B), block 384 = 12 heads x 32 lanes; lane handles 2 channels
__global__ __launch_bounds__(384)
void k_deform(const float* __restrict__ offatt,          // [B*Lq,144]
              const float* __restrict__ refp,            // [B,Lq,1,2]
              const unsigned short* __restrict__ value,  // [B*HW, 768] bf16
              unsigned short* __restrict__ samp,         // [B*Lq, 768] bf16
              const int* __restrict__ Hp, const int* __restrict__ Wp) {
    const int H = Hp[0], W = Wp[0];
    const int q = blockIdx.x, b = blockIdx.y;
    const int t = threadIdx.x;
    const int hh = t >> 5, lane = t & 31;
    const size_t row = (size_t)b * LQ + q;
    const float* oa = offatt + row * 144;

    // softmax over the 4 points of this head
    float lg[4];
    for (int p = 0; p < 4; ++p) lg[p] = oa[96 + hh * 4 + p];
    float mx = fmaxf(fmaxf(lg[0], lg[1]), fmaxf(lg[2], lg[3]));
    float sum = 0.f;
    for (int p = 0; p < 4; ++p) { lg[p] = __expf(lg[p] - mx); sum += lg[p]; }
    float inv = 1.f / sum;

    const float rx = refp[row * 2 + 0], ry = refp[row * 2 + 1];
    const int c0 = lane * 2;
    const unsigned short* vb = value + (size_t)b * H * W * CDIM + hh * HD + c0;

    float a0 = 0.f, a1 = 0.f;
    for (int p = 0; p < 4; ++p) {
        float ox = oa[hh * 8 + p * 2], oy = oa[hh * 8 + p * 2 + 1];
        float x = (rx + ox / (float)W) * (float)W - 0.5f;
        float y = (ry + oy / (float)H) * (float)H - 0.5f;
        float xf = floorf(x), yf = floorf(y);
        int   x0 = (int)xf,  y0 = (int)yf;
        float wx1 = x - xf,  wy1 = y - yf;
        float wx0 = 1.f - wx1, wy0 = 1.f - wy1;
        float aw = lg[p] * inv;
        for (int ty = 0; ty < 2; ++ty)
            for (int tx = 0; tx < 2; ++tx) {
                int xi = x0 + tx, yi = y0 + ty;
                if (xi < 0 || xi >= W || yi < 0 || yi >= H) continue;
                float wgt = aw * (tx ? wx1 : wx0) * (ty ? wy1 : wy0);
                unsigned int pair = *(const unsigned int*)(vb + (size_t)(yi * W + xi) * CDIM);
                a0 += wgt * bf2f((unsigned short)(pair & 0xFFFFu));
                a1 += wgt * bf2f((unsigned short)(pair >> 16));
            }
    }
    unsigned short* sp = samp + row * CDIM + hh * HD + c0;
    sp[0] = f2bf(a0);
    sp[1] = f2bf(a1);
}

// ---------------------------------------------------------------- launcher
extern "C" void kernel_launch(void* const* d_in, const int* in_sizes, int n_in,
                              void* d_out, int out_size, void* d_ws, size_t ws_size,
                              hipStream_t stream) {
    const float* query = (const float*)d_in[0];
    const float* refp  = (const float*)d_in[1];
    const float* feat  = (const float*)d_in[2];
    const int*   Hp    = (const int*)d_in[5];
    const int*   Wp    = (const int*)d_in[6];
    const float* qn_g  = (const float*)d_in[7];
    const float* qn_b  = (const float*)d_in[8];
    const float* fn_g  = (const float*)d_in[9];
    const float* fn_b  = (const float*)d_in[10];
    const float* Wv    = (const float*)d_in[11];
    const float* bv    = (const float*)d_in[12];
    const float* Woff  = (const float*)d_in[13];
    const float* boff  = (const float*)d_in[14];
    const float* Watt  = (const float*)d_in[15];
    const float* batt  = (const float*)d_in[16];
    const float* Wout  = (const float*)d_in[17];
    const float* bout  = (const float*)d_in[18];
    const float* ffn_g = (const float*)d_in[19];
    const float* ffn_b = (const float*)d_in[20];
    const float* W1    = (const float*)d_in[21];
    const float* b1    = (const float*)d_in[22];
    const float* W2    = (const float*)d_in[23];
    const float* b2    = (const float*)d_in[24];
    float* outp = (float*)d_out;

    const size_t M = (size_t)BATCH * LQ;  // 16384
    unsigned char* w = (unsigned char*)d_ws;
    auto alloc = [&](size_t bytes) {
        unsigned char* p = w;
        w += (bytes + 255) & ~(size_t)255;
        return p;
    };
    unsigned short* qnbf   = (unsigned short*)alloc(M * CDIM * 2);  // qn, later FFN-LN
    unsigned short* fnbf   = (unsigned short*)alloc(M * CDIM * 2);  // fn, later samp
    unsigned short* valbf  = (unsigned short*)alloc(M * CDIM * 2);
    float*          offatt = (float*)        alloc(M * 144 * 4);
    unsigned short* hidbf  = (unsigned short*)alloc(M * 192 * 2);
    unsigned short* WvT    = (unsigned short*)alloc(768ull * 768 * 2);   // [N=768,K=768]
    unsigned short* WoutT  = (unsigned short*)alloc(768ull * 768 * 2);   // [768,768]
    unsigned short* WoaT   = (unsigned short*)alloc(144ull * 768 * 2);   // [144,768]
    unsigned short* W1T    = (unsigned short*)alloc(192ull * 768 * 2);   // [192,768]
    unsigned short* W2T    = (unsigned short*)alloc(768ull * 192 * 2);   // [768,192]
    float*          boa    = (float*)        alloc(144 * 4);

    // weight transpose-conversion / packing (tiny, once per call)
    k_cvt_transpose<<<(768 * 768 + 255) / 256, 256, 0, stream>>>(Wv, WvT, 768, 768);
    k_cvt_transpose<<<(768 * 768 + 255) / 256, 256, 0, stream>>>(Wout, WoutT, 768, 768);
    k_cvt_transpose<<<(768 * 192 + 255) / 256, 256, 0, stream>>>(W1, W1T, 768, 192);
    k_cvt_transpose<<<(192 * 768 + 255) / 256, 256, 0, stream>>>(W2, W2T, 192, 768);
    k_pack_offatt_w<<<(768 * 144 + 255) / 256, 256, 0, stream>>>(Woff, Watt, WoaT);
    k_pack_offatt_b<<<1, 144, 0, stream>>>(boff, batt, boa);

    // layernorms -> bf16
    k_layernorm_bf16<<<(int)M, 256, 0, stream>>>(query, qn_g, qn_b, qnbf);
    k_layernorm_bf16<<<(int)M, 256, 0, stream>>>(feat,  fn_g, fn_b, fnbf);

    dim3 blk(256);
    dim3 gFull(768 / 128, (int)(M / 128));        // N=768
    dim3 gOA((144 + 127) / 128, (int)(M / 128));  // N=144
    dim3 gHid((192 + 127) / 128, (int)(M / 128)); // N=192

    // value = fn @ Wv + bv  (bf16 out)
    k_gemm_bf16<true, false, false, true><<<gFull, blk, 0, stream>>>(
        fnbf, WvT, bv, nullptr, valbf, (int)M, 768, 768);
    // [off | att_logits] = qn @ [Woff|Watt] + [boff|batt]  (f32 out)
    k_gemm_bf16<false, false, false, false><<<gOA, blk, 0, stream>>>(
        qnbf, WoaT, boa, nullptr, offatt, (int)M, 144, 768);
    // deformable sampling -> samp (reuses fn buffer)
    dim3 gDef(LQ, BATCH);
    k_deform<<<gDef, 384, 0, stream>>>(offatt, refp, valbf, fnbf, Hp, Wp);
    // query2 = query + samp @ Wout + bout  (f32 -> d_out)
    k_gemm_bf16<false, false, true, true><<<gFull, blk, 0, stream>>>(
        fnbf, WoutT, bout, query, outp, (int)M, 768, 768);
    // h = LN(query2)
    k_layernorm_bf16<<<(int)M, 256, 0, stream>>>(outp, ffn_g, ffn_b, qnbf);
    // hid = gelu(h @ W1 + b1)  (bf16)
    k_gemm_bf16<true, true, false, false><<<gHid, blk, 0, stream>>>(
        qnbf, W1T, b1, nullptr, hidbf, (int)M, 192, 768);
    // out = query2 + hid @ W2 + b2  (in-place residual on d_out, K=192)
    k_gemm_bf16<false, false, true, true><<<gFull, blk, 0, stream>>>(
        hidbf, W2T, b2, outp, outp, (int)M, 768, 192);
}
